// GraphConvolution_46033459479198
// MI455X (gfx1250) — matrix operand order repaired and checked by reading, not compiled, not run on hardware
//
#include <hip/hip_runtime.h>

typedef float  v2f  __attribute__((ext_vector_type(2)));
typedef float  v4f  __attribute__((ext_vector_type(4)));
typedef float  v8f  __attribute__((ext_vector_type(8)));
typedef __bf16 v4bf __attribute__((ext_vector_type(4)));
typedef __bf16 v8bf __attribute__((ext_vector_type(8)));
typedef __bf16 v16bf __attribute__((ext_vector_type(16)));

#define N_NODES 100000
#define N_EDGES 1600000
#define D_IN    256
#define D_OUT   64

// LDS bf16 row stride: 264 elems = 528 B = 132 dwords; 132 % 64 == 4 ->
// b128 fragment reads hit each bank at most twice (optimal for 512B/wave),
// and 528 is 16B-aligned so ds_load_b128 alignment holds for all offsets used.
#define XS_STRIDE 264
#define XS_PART   (32 * XS_STRIDE)     // elems per (hi/lo) part

// W fragment buffer: [colTile(4)][chunk(8)][lane(32)][16 bf16] = 16384 elems/part
#define WFRAG_ELEMS (4 * 8 * 32 * 16)
#define SUPPORT_BYTES ((size_t)N_NODES * D_OUT * sizeof(float))   // 25.6 MB

__device__ __forceinline__ v16bf cat16(v8bf a, v8bf b) {
  return __builtin_shufflevector(a, b, 0,1,2,3,4,5,6,7,8,9,10,11,12,13,14,15);
}

// ---------------------------------------------------------------------------
// Pre-split W (256x64 f32) into bf16 hi/lo, stored in per-lane B-fragment
// order: idx = ((ct*8 + chunk)*32 + lane)*16 + j, where the 16 bf16 for a
// lane are K = chunk*32 + 16*(lane>>4) + j at column N = ct*16 + (lane&15).
// ---------------------------------------------------------------------------
__global__ __launch_bounds__(256) void gcn_prep_w(const float* __restrict__ w,
                                                  __bf16* __restrict__ wh,
                                                  __bf16* __restrict__ wl) {
  int idx = blockIdx.x * 256 + threadIdx.x;        // 0..16383
  int j    = idx & 15;
  int lane = (idx >> 4) & 31;
  int c    = (idx >> 9) & 7;
  int ct   = idx >> 12;
  int k = c * 32 + (lane >> 4) * 16 + j;
  int n = ct * 16 + (lane & 15);
  float v = w[k * D_OUT + n];
  __bf16 h = (__bf16)v;
  __bf16 l = (__bf16)(v - (float)h);
  wh[idx] = h;
  wl[idx] = l;
}

// ---------------------------------------------------------------------------
// GEMM: support = x @ W via bf16x3 split-precision WMMA (16x16x32 bf16).
// 256 threads = 8 waves; block tile 32 rows x 64 cols; wave tile 16x16.
// ---------------------------------------------------------------------------
__global__ __launch_bounds__(256) void gcn_gemm(const float* __restrict__ x,
                                                const __bf16* __restrict__ wh,
                                                const __bf16* __restrict__ wl,
                                                float* __restrict__ support) {
  __shared__ __bf16 xs[2 * XS_PART];               // [part hi/lo][32][264]
  const int tid  = threadIdx.x;
  const int lane = tid & 31;
  const int wave = tid >> 5;
  const int rowBase = blockIdx.x * 32;

  // Load 32x256 f32 tile, split to bf16 hi/lo, store to LDS (coalesced).
#pragma unroll
  for (int i = 0; i < 8; ++i) {
    int idx = i * 256 + tid;                       // 0..2047 float4 slots
    int r   = idx >> 6;
    int c4  = idx & 63;
    v4f v = *(const v4f*)(x + (size_t)(rowBase + r) * D_IN + c4 * 4);
    v4bf hv, lv;
#pragma unroll
    for (int e = 0; e < 4; ++e) {
      __bf16 h = (__bf16)v[e];
      hv[e] = h;
      lv[e] = (__bf16)(v[e] - (float)h);
    }
    *(v4bf*)(xs + r * XS_STRIDE + c4 * 4)           = hv;
    *(v4bf*)(xs + XS_PART + r * XS_STRIDE + c4 * 4) = lv;
  }
  __syncthreads();

  const int rowTile = (wave & 1) * 16;             // 0 / 16
  const int colTile = wave >> 1;                   // 0..3
  const int g  = lane >> 4;                        // half-wave
  const int lo = lane & 15;

  // A fragment (16x32 bf16): VGPR0-7 hold K = 16g+0..15
  const __bf16* aH = xs + (rowTile + lo) * XS_STRIDE + 16 * g;
  const __bf16* aL = aH + XS_PART;
  // B fragment: lane's 16 bf16 contiguous at wfrag[((ct*8+c)*32 + lane)*16]
  const __bf16* bH = wh + ((size_t)(colTile * 8) * 32 + lane) * 16;
  const __bf16* bL = wl + ((size_t)(colTile * 8) * 32 + lane) * 16;

  v8f accHH = {}, accHL = {}, accLH = {};
#pragma unroll
  for (int c = 0; c < 8; ++c) {
    const int ko = c * 32;                          // K base in LDS
    const int bo = c * 512;                         // 32 lanes * 16 elems
    v16bf Ah = cat16(*(const v8bf*)(aH + ko), *(const v8bf*)(aH + ko + 8));
    v16bf Al = cat16(*(const v8bf*)(aL + ko), *(const v8bf*)(aL + ko + 8));
    v16bf Bh = cat16(*(const v8bf*)(bH + bo), *(const v8bf*)(bH + bo + 8));
    v16bf Bl = cat16(*(const v8bf*)(bL + bo), *(const v8bf*)(bL + bo + 8));
    accHH = __builtin_amdgcn_wmma_f32_16x16x32_bf16(false, Ah, false, Bh,
                                                    (short)0, accHH, false, false);
    accHL = __builtin_amdgcn_wmma_f32_16x16x32_bf16(false, Ah, false, Bl,
                                                    (short)0, accHL, false, false);
    accLH = __builtin_amdgcn_wmma_f32_16x16x32_bf16(false, Al, false, Bh,
                                                    (short)0, accLH, false, false);
  }
  v8f acc = accHH + accHL + accLH;

  // C/D layout: VGPR r -> row M = r + 8*g, col N = lo.
  float* outp = support + (size_t)(rowBase + rowTile + 8 * g) * D_OUT
              + colTile * 16 + lo;
#pragma unroll
  for (int r = 0; r < 8; ++r)
    outp[(size_t)r * D_OUT] = acc[r];
}

// ---------------------------------------------------------------------------
// out[i] = bias[i % 64]
// ---------------------------------------------------------------------------
__global__ __launch_bounds__(256) void gcn_init_out(float* __restrict__ out,
                                                    const float* __restrict__ bias) {
  int i = blockIdx.x * 256 + threadIdx.x;
  if (i < N_NODES * D_OUT) out[i] = bias[i & (D_OUT - 1)];
}

// ---------------------------------------------------------------------------
// SpMM with sorted rows: register run-length reduction, flush on row change.
// blockDim = (64,4): x = feature, y = edge segment. 512 edges per segment.
// ---------------------------------------------------------------------------
#define SEG_EDGES 512

__global__ __launch_bounds__(256) void gcn_spmm(const float* __restrict__ support,
                                                const float* __restrict__ vals,
                                                const int* __restrict__ rows,
                                                const int* __restrict__ cols,
                                                float* __restrict__ out) {
  const int f   = threadIdx.x;                     // feature 0..63
  const int seg = blockIdx.x * 4 + threadIdx.y;
  long e0 = (long)seg * SEG_EDGES;
  if (e0 >= N_EDGES) return;
  long e1 = e0 + SEG_EDGES;
  if (e1 > N_EDGES) e1 = N_EDGES;

  int   curRow = rows[e0];
  float acc    = 0.0f;

  for (long e = e0; e < e1; ++e) {
    int r = rows[e];
    if (r != curRow) {
      atomicAdd(out + (size_t)curRow * D_OUT + f, acc);
      acc = 0.0f;
      curRow = r;
    }
    acc = fmaf(vals[e], support[(size_t)cols[e] * D_OUT + f], acc);
  }
  atomicAdd(out + (size_t)curRow * D_OUT + f, acc);
}

// ---------------------------------------------------------------------------
extern "C" void kernel_launch(void* const* d_in, const int* in_sizes, int n_in,
                              void* d_out, int out_size, void* d_ws, size_t ws_size,
                              hipStream_t stream) {
  const float* x        = (const float*)d_in[0];
  const float* w        = (const float*)d_in[1];
  const float* bias     = (const float*)d_in[2];
  const float* adj_vals = (const float*)d_in[3];
  const int*   adj_row  = (const int*)d_in[4];
  const int*   adj_col  = (const int*)d_in[5];
  float*       out      = (float*)d_out;

  float*  support = (float*)d_ws;
  __bf16* wfragH  = (__bf16*)((char*)d_ws + SUPPORT_BYTES);
  __bf16* wfragL  = wfragH + WFRAG_ELEMS;

  // 0) split W into bf16 hi/lo fragments (per-lane layout)
  gcn_prep_w<<<WFRAG_ELEMS / 256, 256, 0, stream>>>(w, wfragH, wfragL);

  // 1) support = x @ W   (bf16x3 WMMA, fp32 accumulate)
  gcn_gemm<<<N_NODES / 32, 256, 0, stream>>>(x, wfragH, wfragL, support);

  // 2) out = bias (broadcast)
  gcn_init_out<<<(N_NODES * D_OUT) / 256, 256, 0, stream>>>(out, bias);

  // 3) out[r] += vals[e] * support[c]  (sorted-row segmented reduction)
  int nseg    = (N_EDGES + SEG_EDGES - 1) / SEG_EDGES;   // 3125
  int nblocks = (nseg + 3) / 4;                          // 782
  gcn_spmm<<<nblocks, dim3(64, 4), 0, stream>>>(support, adj_vals, adj_row, adj_col, out);
}
